// Attention_4681514353037
// MI455X (gfx1250) — matrix operand order repaired and checked by reading, not compiled, not run on hardware
//
#include <hip/hip_runtime.h>
#include <hip/hip_bf16.h>

// ---------------------------------------------------------------------------
// Problem constants (reference: B=8, S=1024, H=16, HD=64, D=1024)
// ---------------------------------------------------------------------------
#define BB   8
#define SS   1024
#define HH   16
#define HDD  64
#define DD   1024
#define MM   (BB * SS)      // 8192 rows

typedef __bf16 bf16_t;
typedef __attribute__((ext_vector_type(16))) __bf16 v16bf;
typedef __attribute__((ext_vector_type(8)))  __bf16 v8bf;
typedef __attribute__((ext_vector_type(8)))  float  v8f;

// Build a 16-element bf16 fragment from two contiguous 16B chunks.
// Matches CDNA5 16-bit A/B VGPR layouts (elements 0..7 then 8..15).
__device__ __forceinline__ v16bf ld16(const bf16_t* p0, const bf16_t* p1) {
    v8bf a = *(const v8bf*)p0;
    v8bf b = *(const v8bf*)p1;
    v16bf r;
#pragma unroll
    for (int i = 0; i < 8; ++i) { r[i] = a[i]; r[i + 8] = b[i]; }
    return r;
}

__device__ __forceinline__ v16bf ld16s(const bf16_t* p, int ld) {
    v16bf r;
#pragma unroll
    for (int i = 0; i < 8; ++i) { r[i] = p[i * ld]; r[i + 8] = p[(i + 16) * ld]; }
    return r;
}

__device__ __forceinline__ v8f wmma_bf16(v16bf a, v16bf b, v8f c) {
    return __builtin_amdgcn_wmma_f32_16x16x32_bf16(
        false, a, false, b, (short)0, c, false, false);
}

// ---------------------------------------------------------------------------
// CDNA5 async global->LDS copy (ASYNCcnt-tracked, bypasses VGPRs).
// saddr form: 64-bit uniform base in SGPRs + per-lane 32-bit byte offset.
// LDS destination address = low 32 bits of the generic pointer.
// ---------------------------------------------------------------------------
__device__ __forceinline__ void async_b128(const void* lds_dst,
                                           const void* gbase,
                                           unsigned byte_off) {
    unsigned lds_off = (unsigned)(unsigned long long)lds_dst;
    asm volatile("global_load_async_to_lds_b128 %0, %1, %2"
                 :: "v"(lds_off), "v"(byte_off), "s"(gbase)
                 : "memory");
}
__device__ __forceinline__ void wait_async0() {
    asm volatile("s_wait_asynccnt 0x0" ::: "memory");
}

// ---------------------------------------------------------------------------
// fp32 -> bf16 conversion (4 elems / thread, packed 8B store)
// ---------------------------------------------------------------------------
__global__ void cvt_kernel(const float* __restrict__ in, bf16_t* __restrict__ out, int n4) {
    int i = blockIdx.x * 256 + threadIdx.x;
    if (i < n4) {
        float4 f = ((const float4*)in)[i];
        union { bf16_t h[4]; uint2 u; } r;
        r.h[0] = (bf16_t)f.x; r.h[1] = (bf16_t)f.y;
        r.h[2] = (bf16_t)f.z; r.h[3] = (bf16_t)f.w;
        ((uint2*)out)[i] = r.u;
    }
}

// ---------------------------------------------------------------------------
// GEMM main loop: block tile 128x128, 8 waves as 4(M) x 2(N), wave tile 32x64.
// Double-buffered LDS filled by async-to-LDS; one barrier per k-step.
// ---------------------------------------------------------------------------
__device__ __forceinline__ void gemm_issue(
    const bf16_t* A, const bf16_t* Bw, int mBase, int nBase, int k0,
    bf16_t (*sA)[32], bf16_t (*sB)[128], int tid)
{
#pragma unroll
    for (int rep = 0; rep < 2; ++rep) {
        int i = tid + rep * 256;
        {   // A tile 128x32 (512 x 16B)
            int r = i >> 2, c = (i & 3) << 3;
            async_b128(&sA[r][c], A, (unsigned)(((mBase + r) * DD + k0 + c) * 2));
        }
        {   // B tile 32x128 (512 x 16B)
            int r = i >> 4, c = (i & 15) << 3;
            async_b128(&sB[r][c], Bw, (unsigned)(((k0 + r) * DD + nBase + c) * 2));
        }
    }
}

__device__ __forceinline__ void gemm_mainloop(
    const bf16_t* __restrict__ A, const bf16_t* __restrict__ Bw,
    int mBase, int nBase,
    bf16_t (*sA0)[32], bf16_t (*sA1)[32],
    bf16_t (*sB0)[128], bf16_t (*sB1)[128],
    v8f acc[2][4])
{
    const int tid  = threadIdx.x;
    const int lane = tid & 31;
    const int wv   = tid >> 5;
    const int mw   = (wv & 3) << 5;             // 0,32,64,96
    const int nw   = (wv >> 2) << 6;            // 0,64
    const int ql   = lane & 15;
    const int ab   = (lane >> 4) << 3;          // 0 or 8 (K split per lane half)

    gemm_issue(A, Bw, mBase, nBase, 0, sA0, sB0, tid);

    for (int ks = 0; ks < DD / 32; ++ks) {
        bf16_t (*cA)[32]  = (ks & 1) ? sA1 : sA0;
        bf16_t (*cB)[128] = (ks & 1) ? sB1 : sB0;
        wait_async0();           // our tile's async fills have landed in LDS
        __syncthreads();         // everyone's fills visible; prev buffer free
        if (ks + 1 < DD / 32)
            gemm_issue(A, Bw, mBase, nBase, (ks + 1) * 32,
                       (ks & 1) ? sA0 : sA1, (ks & 1) ? sB0 : sB1, tid);

        v16bf af0 = ld16(&cA[mw + ql][ab],      &cA[mw + ql][ab + 16]);
        v16bf af1 = ld16(&cA[mw + 16 + ql][ab], &cA[mw + 16 + ql][ab + 16]);
#pragma unroll
        for (int j = 0; j < 4; ++j) {
            v16bf bf = ld16s(&cB[ab][nw + j * 16 + ql], 128);
            acc[0][j] = wmma_bf16(af0, bf, acc[0][j]);
            acc[1][j] = wmma_bf16(af1, bf, acc[1][j]);
        }
    }
}

// ---------------------------------------------------------------------------
// QKV projection: out = (X @ W + bias), stored bf16 in attention-friendly layout
//   mode 0 (Q): scaled by 1/sqrt(HD), layout [bh][s][hd]
//   mode 1 (K): layout [bh][hd][s]   (transposed for score B-fragments)
//   mode 2 (V): layout [bh][s][hd]
// ---------------------------------------------------------------------------
__global__ __launch_bounds__(256) void gemm_qkv_kernel(
    const bf16_t* __restrict__ Xh, const bf16_t* __restrict__ Wh,
    const float* __restrict__ bias, bf16_t* __restrict__ out, int mode)
{
    __shared__ __align__(16) bf16_t sA[2][128][32];
    __shared__ __align__(16) bf16_t sB[2][32][128];

    v8f acc[2][4] = {};
    const int mBase = blockIdx.y * 128;
    const int nBase = blockIdx.x * 128;
    gemm_mainloop(Xh, Wh, mBase, nBase, sA[0], sA[1], sB[0], sB[1], acc);

    const int lane = threadIdx.x & 31;
    const int wv   = threadIdx.x >> 5;
    const int mw   = (wv & 3) << 5;
    const int nw   = (wv >> 2) << 6;
    const int ql   = lane & 15;
    const int ab   = (lane >> 4) << 3;
#pragma unroll
    for (int mi = 0; mi < 2; ++mi) {
        int rb = mBase + mw + mi * 16 + ab;
#pragma unroll
        for (int j = 0; j < 4; ++j) {
            int col = nBase + nw + j * 16 + ql;
            int h = col >> 6, hd = col & 63;
            float bv = bias[col];
#pragma unroll
            for (int r = 0; r < 8; ++r) {
                int row = rb + r;
                int b = row >> 10, s = row & 1023;
                float v = acc[mi][j][r] + bv;
                if (mode == 0) v *= 0.125f;          // 1/sqrt(64)
                int bh = b * HH + h;
                size_t idx = (mode == 1)
                    ? ((size_t)bh * HDD + hd) * SS + s
                    : ((size_t)bh * SS + s) * HDD + hd;
                out[idx] = (bf16_t)v;
            }
        }
    }
}

// ---------------------------------------------------------------------------
// Flash attention: per (bh, 128-query tile). Online softmax over 64-key tiles.
// ---------------------------------------------------------------------------
__global__ __launch_bounds__(256) void attn_kernel(
    const bf16_t* __restrict__ Qg, const bf16_t* __restrict__ Ktg,
    const bf16_t* __restrict__ Vg, const int* __restrict__ maskg,
    bf16_t* __restrict__ ctx)
{
    __shared__ __align__(16) bf16_t sQ[128][64];
    __shared__ __align__(16) bf16_t sKt[64][64];   // [hd][key]
    __shared__ __align__(16) bf16_t sV[64][64];    // [key][hd]
    __shared__ __align__(16) float  sM[64];
    __shared__ __align__(16) bf16_t sP[8][16][64]; // per-wave P round-trip

    const int tid  = threadIdx.x;
    const int lane = tid & 31;
    const int wv   = tid >> 5;
    const int bh   = blockIdx.y;
    const int b    = bh >> 4;
    const int h    = bh & 15;
    const int qBase = blockIdx.x * 128;

    const bf16_t* QBase  = Qg  + ((size_t)bh * SS + qBase) * HDD;
    const bf16_t* KtBase = Ktg + (size_t)bh * HDD * SS;
    const bf16_t* VBase  = Vg  + (size_t)bh * SS * HDD;

    // Stage Q tile (contiguous 16 KB) via async-to-LDS
#pragma unroll
    for (int rep = 0; rep < 4; ++rep) {
        int i = tid + rep * 256;
        async_b128(((bf16_t*)sQ) + i * 8, QBase, (unsigned)(i * 16));
    }
    wait_async0();
    __syncthreads();

    const int ql = lane & 15;
    const int ab = (lane >> 4) << 3;               // 0 or 8

    v16bf qf0 = ld16(&sQ[wv * 16 + ql][ab],      &sQ[wv * 16 + ql][ab + 16]);
    v16bf qf1 = ld16(&sQ[wv * 16 + ql][32 + ab], &sQ[wv * 16 + ql][32 + ab + 16]);

    float mrow[8], lrow[8];
    v8f oacc[4] = {};
#pragma unroll
    for (int r = 0; r < 8; ++r) { mrow[r] = -3.0e38f; lrow[r] = 0.f; }

    for (int kt = 0; kt < SS; kt += 64) {
        __syncthreads();   // previous tile's reads complete everywhere
        // K^T tile [hd][key] and V tile [key][hd], async-to-LDS
#pragma unroll
        for (int rep = 0; rep < 2; ++rep) {
            int i = tid + rep * 256;
            int hd = i >> 3, c = (i & 7) << 3;
            async_b128(&sKt[hd][c], KtBase, (unsigned)((hd * SS + kt + c) * 2));
            async_b128(((bf16_t*)sV) + i * 8, VBase, (unsigned)(kt * HDD * 2 + i * 16));
        }
        if (tid < 64) sM[tid] = (float)maskg[b * SS + kt + tid] * -10000.0f;
        wait_async0();
        __syncthreads();

        // scores = Q(16x64) @ K^T(64x64)
        v8f sc[4] = {};
#pragma unroll
        for (int j = 0; j < 4; ++j) {
            v16bf b0 = ld16s(&sKt[ab][j * 16 + ql], 64);
            sc[j] = wmma_bf16(qf0, b0, sc[j]);
            v16bf b1 = ld16s(&sKt[32 + ab][j * 16 + ql], 64);
            sc[j] = wmma_bf16(qf1, b1, sc[j]);
        }
        // padding mask (per key column)
#pragma unroll
        for (int j = 0; j < 4; ++j) {
            float mval = sM[j * 16 + ql];
#pragma unroll
            for (int r = 0; r < 8; ++r) sc[j][r] += mval;
        }
        // online softmax: rows live across 16-lane halves -> shfl_xor 1,2,4,8
#pragma unroll
        for (int r = 0; r < 8; ++r) {
            float tm = fmaxf(fmaxf(sc[0][r], sc[1][r]), fmaxf(sc[2][r], sc[3][r]));
#pragma unroll
            for (int off = 8; off >= 1; off >>= 1)
                tm = fmaxf(tm, __shfl_xor(tm, off, 32));
            float mnew  = fmaxf(mrow[r], tm);
            float scale = __expf(mrow[r] - mnew);
            mrow[r] = mnew;
            lrow[r] *= scale;
#pragma unroll
            for (int j = 0; j < 4; ++j) oacc[j][r] *= scale;
            float rs = 0.f;
#pragma unroll
            for (int j = 0; j < 4; ++j) {
                float e = __expf(sc[j][r] - mnew);
                sc[j][r] = e;
                rs += e;
            }
#pragma unroll
            for (int off = 8; off >= 1; off >>= 1)
                rs += __shfl_xor(rs, off, 32);
            lrow[r] += rs;
        }
        // P: C-layout -> A-layout via wave-private LDS round-trip
#pragma unroll
        for (int j = 0; j < 4; ++j)
#pragma unroll
            for (int r = 0; r < 8; ++r)
                sP[wv][r + ab][j * 16 + ql] = (bf16_t)sc[j][r];

        v16bf pf0 = ld16(&sP[wv][ql][ab],      &sP[wv][ql][ab + 16]);
        v16bf pf1 = ld16(&sP[wv][ql][32 + ab], &sP[wv][ql][32 + ab + 16]);

        // ctx += P(16x64) @ V(64x64)
#pragma unroll
        for (int j = 0; j < 4; ++j) {
            v16bf b0 = ld16s(&sV[ab][j * 16 + ql], 64);
            oacc[j] = wmma_bf16(pf0, b0, oacc[j]);
            v16bf b1 = ld16s(&sV[32 + ab][j * 16 + ql], 64);
            oacc[j] = wmma_bf16(pf1, b1, oacc[j]);
        }
    }

    // normalize + store ctx as bf16 in [B*S, D] layout for the output GEMM
#pragma unroll
    for (int j = 0; j < 4; ++j) {
        int hd = j * 16 + ql;
#pragma unroll
        for (int r = 0; r < 8; ++r) {
            int s = qBase + wv * 16 + ab + r;
            float v = oacc[j][r] / lrow[r];
            ctx[((size_t)(b * SS + s)) * DD + h * HDD + hd] = (bf16_t)v;
        }
    }
}

// ---------------------------------------------------------------------------
// Output projection + bias + residual: out_f32 = x + ctx @ Wo + bo
// ---------------------------------------------------------------------------
__global__ __launch_bounds__(256) void gemm_out_kernel(
    const bf16_t* __restrict__ Ch, const bf16_t* __restrict__ Woh,
    const float* __restrict__ bo, const float* __restrict__ x,
    float* __restrict__ out)
{
    __shared__ __align__(16) bf16_t sA[2][128][32];
    __shared__ __align__(16) bf16_t sB[2][32][128];

    v8f acc[2][4] = {};
    const int mBase = blockIdx.y * 128;
    const int nBase = blockIdx.x * 128;
    gemm_mainloop(Ch, Woh, mBase, nBase, sA[0], sA[1], sB[0], sB[1], acc);

    const int lane = threadIdx.x & 31;
    const int wv   = threadIdx.x >> 5;
    const int mw   = (wv & 3) << 5;
    const int nw   = (wv >> 2) << 6;
    const int ql   = lane & 15;
    const int ab   = (lane >> 4) << 3;
#pragma unroll
    for (int mi = 0; mi < 2; ++mi) {
        int rb = mBase + mw + mi * 16 + ab;
#pragma unroll
        for (int j = 0; j < 4; ++j) {
            int col = nBase + nw + j * 16 + ql;
            float bv = bo[col];
#pragma unroll
            for (int r = 0; r < 8; ++r) {
                size_t idx = (size_t)(rb + r) * DD + col;
                out[idx] = acc[mi][j][r] + bv + x[idx];
            }
        }
    }
}

// ---------------------------------------------------------------------------
// In-place row LayerNorm over D=1024 (one block per row)
// ---------------------------------------------------------------------------
__global__ __launch_bounds__(256) void ln_kernel(
    float* __restrict__ buf, const float* __restrict__ gamma,
    const float* __restrict__ beta)
{
    __shared__ float s1[8], s2[8];
    const int tid = threadIdx.x;
    float* p = buf + (size_t)blockIdx.x * DD;

    float x0 = p[tid], x1 = p[tid + 256], x2 = p[tid + 512], x3 = p[tid + 768];
    float sum = x0 + x1 + x2 + x3;
    float sq  = x0 * x0 + x1 * x1 + x2 * x2 + x3 * x3;
#pragma unroll
    for (int off = 16; off >= 1; off >>= 1) {
        sum += __shfl_xor(sum, off, 32);
        sq  += __shfl_xor(sq,  off, 32);
    }
    if ((tid & 31) == 0) { s1[tid >> 5] = sum; s2[tid >> 5] = sq; }
    __syncthreads();
    sum = 0.f; sq = 0.f;
#pragma unroll
    for (int i = 0; i < 8; ++i) { sum += s1[i]; sq += s2[i]; }
    float mu  = sum * (1.0f / DD);
    float var = sq * (1.0f / DD) - mu * mu;
    float rs  = rsqrtf(var + 1e-6f);
    p[tid]       = (x0 - mu) * rs * gamma[tid]       + beta[tid];
    p[tid + 256] = (x1 - mu) * rs * gamma[tid + 256] + beta[tid + 256];
    p[tid + 512] = (x2 - mu) * rs * gamma[tid + 512] + beta[tid + 512];
    p[tid + 768] = (x3 - mu) * rs * gamma[tid + 768] + beta[tid + 768];
}

// ---------------------------------------------------------------------------
// Launcher
// ---------------------------------------------------------------------------
extern "C" void kernel_launch(void* const* d_in, const int* in_sizes, int n_in,
                              void* d_out, int out_size, void* d_ws, size_t ws_size,
                              hipStream_t stream) {
    const float* x     = (const float*)d_in[0];
    const int*   mask  = (const int*)  d_in[1];
    const float* wq    = (const float*)d_in[2];
    const float* bq    = (const float*)d_in[3];
    const float* wk    = (const float*)d_in[4];
    const float* bk    = (const float*)d_in[5];
    const float* wv    = (const float*)d_in[6];
    const float* bv    = (const float*)d_in[7];
    const float* wo    = (const float*)d_in[8];
    const float* bo    = (const float*)d_in[9];
    const float* gamma = (const float*)d_in[10];
    const float* beta  = (const float*)d_in[11];
    float* out = (float*)d_out;

    char* ws = (char*)d_ws;
    const size_t MB = 1u << 20;
    bf16_t* Xh  = (bf16_t*)(ws + 0 * MB);    // 16 MB: x bf16 [8192,1024]
    bf16_t* Wqh = (bf16_t*)(ws + 16 * MB);   //  2 MB each
    bf16_t* Wkh = (bf16_t*)(ws + 18 * MB);
    bf16_t* Wvh = (bf16_t*)(ws + 20 * MB);
    bf16_t* Woh = (bf16_t*)(ws + 22 * MB);
    bf16_t* Qh  = (bf16_t*)(ws + 24 * MB);   // 16 MB [bh][s][hd] (pre-scaled)
    bf16_t* Kth = (bf16_t*)(ws + 40 * MB);   // 16 MB [bh][hd][s]
    bf16_t* Vh  = (bf16_t*)(ws + 56 * MB);   // 16 MB [bh][s][hd]
    bf16_t* Ch  = (bf16_t*)(ws + 72 * MB);   // 16 MB ctx [8192,1024]

    // fp32 -> bf16 conversions
    cvt_kernel<<<8192, 256, 0, stream>>>(x,  Xh,  (MM * DD) / 4);
    cvt_kernel<<<1024, 256, 0, stream>>>(wq, Wqh, (DD * DD) / 4);
    cvt_kernel<<<1024, 256, 0, stream>>>(wk, Wkh, (DD * DD) / 4);
    cvt_kernel<<<1024, 256, 0, stream>>>(wv, Wvh, (DD * DD) / 4);
    cvt_kernel<<<1024, 256, 0, stream>>>(wo, Woh, (DD * DD) / 4);

    dim3 gg(DD / 128, MM / 128);             // (8, 64)
    gemm_qkv_kernel<<<gg, 256, 0, stream>>>(Xh, Wqh, bq, Qh,  0);
    gemm_qkv_kernel<<<gg, 256, 0, stream>>>(Xh, Wkh, bk, Kth, 1);
    gemm_qkv_kernel<<<gg, 256, 0, stream>>>(Xh, Wvh, bv, Vh,  2);

    attn_kernel<<<dim3(SS / 128, BB * HH), 256, 0, stream>>>(Qh, Kth, Vh, mask, Ch);

    gemm_out_kernel<<<gg, 256, 0, stream>>>(Ch, Woh, bo, x, out);
    ln_kernel<<<MM, 256, 0, stream>>>(out, gamma, beta);
}
